// RelationAwareAttention_16947940950217
// MI455X (gfx1250) — matrix operand order repaired and checked by reading, not compiled, not run on hardware
//
#include <hip/hip_runtime.h>
#include <hip/hip_bf16.h>

// ---------------------------------------------------------------------------
// Relation-aware attention for MI455X (gfx1250, wave32, WMMA bf16)
//   Q = Xmaj (Wq/32), K = Xmin Wk, V = Xmin Wv          (bf16 WMMA GEMMs)
//   S = Q Kt ; B = Q wk^T ; logits = S[i,j] + B[i,table[i,j]]
//   A = softmax(logits) ; Y = (A V) Wo
// Outputs: d_out = [ Y (210x1024 f32) | A (210x210 f32) ]
// ---------------------------------------------------------------------------

typedef __bf16 bf16_t;
typedef __attribute__((ext_vector_type(16))) __bf16 v16bf;
typedef __attribute__((ext_vector_type(8)))  __bf16 v8bf;
typedef __attribute__((ext_vector_type(8)))  float  v8f;

#define NSEQ  210
#define MP    224     // NSEQ padded to multiple of 32
#define DMDIM 2048
#define DKDIM 1024
#define DVDIM 1024
#define NREL  65
#define NRELP 96      // 65 padded to multiple of 32

// ---- operand fragment loads (per ISA 7.12.2 VGPR layouts, wave32) ----------
__device__ __forceinline__ v16bf load_a_frag(const bf16_t* p) {
  // A 16x32 bf16: lane-half h -> elems 0..7 at K+8h, elems 8..15 at K+16+8h
  v8bf lo = *(const v8bf*)(p);
  v8bf hi = *(const v8bf*)(p + 16);
  return __builtin_shufflevector(lo, hi, 0,1,2,3,4,5,6,7,8,9,10,11,12,13,14,15);
}
__device__ __forceinline__ v16bf load_b_frag(const bf16_t* p) {
  // B 32x16 bf16: lane-half h -> elems 0..7 at K+8h, elems 8..15 at K+16+8h
  v8bf lo = *(const v8bf*)(p);
  v8bf hi = *(const v8bf*)(p + 16);
  return __builtin_shufflevector(lo, hi, 0,1,2,3,4,5,6,7,8,9,10,11,12,13,14,15);
}

// ---- tiled WMMA GEMM: C[M,N] = A[M,K] * Bt[N,K]^T --------------------------
// STORE_MODE: 0 = f32 row-major, 1 = bf16 row-major, 2 = bf16 transposed
// BN: number of 16-wide column tiles per wave (wave tile = 32 x 16*BN)
template<int STORE_MODE, int BN>
__global__ __launch_bounds__(256)
void gemm_wmma_bf16(const bf16_t* __restrict__ A, int lda,
                    const bf16_t* __restrict__ Bt, int ldb,
                    void* __restrict__ C, int ldc,
                    int Mt, int Nt, int Kdim, int Mvalid)
{
  int gw = blockIdx.x * 8 + (threadIdx.x >> 5);   // global wave id -> tile
  if (gw >= Mt * Nt) return;
  int tm = gw / Nt, tn = gw % Nt;
  int lane = threadIdx.x & 31;
  int h = lane >> 4;            // lane-half
  int r = lane & 15;            // row (A) / col-of-Bt (B)
  int m0 = tm * 32, n0 = tn * (16 * BN);

  const bf16_t* a0p = A + (size_t)(m0 +      r) * lda + 8 * h;
  const bf16_t* a1p = A + (size_t)(m0 + 16 + r) * lda + 8 * h;
  const bf16_t* bp[BN];
  #pragma unroll
  for (int j = 0; j < BN; ++j)
    bp[j] = Bt + (size_t)(n0 + 16 * j + r) * ldb + 8 * h;

  v8f acc[2][BN];
  #pragma unroll
  for (int i = 0; i < 2; ++i)
    #pragma unroll
    for (int j = 0; j < BN; ++j)
      acc[i][j] = v8f{};

  for (int k0 = 0; k0 < Kdim; k0 += 32) {
    if (k0 + 128 < Kdim) {                 // WGP-scope global_prefetch_b8
      __builtin_prefetch(a0p + k0 + 128, 0, 3);
      __builtin_prefetch(bp[0] + k0 + 128, 0, 3);
      __builtin_prefetch(bp[BN - 1] + k0 + 128, 0, 3);
    }
    v16bf a0 = load_a_frag(a0p + k0);
    v16bf a1 = load_a_frag(a1p + k0);
    #pragma unroll
    for (int j = 0; j < BN; ++j) {
      v16bf b = load_b_frag(bp[j] + k0);
      acc[0][j] = __builtin_amdgcn_wmma_f32_16x16x32_bf16(false, a0, false, b, (short)0, acc[0][j], false, false);
      acc[1][j] = __builtin_amdgcn_wmma_f32_16x16x32_bf16(false, a1, false, b, (short)0, acc[1][j], false, false);
    }
  }

  // C/D layout: VGPR e, lanes0-15 -> M=e, lanes16-31 -> M=e+8 ; N = lane&15
  #pragma unroll
  for (int i = 0; i < 2; ++i) {
    #pragma unroll
    for (int j = 0; j < BN; ++j) {
      int n = n0 + 16 * j + r;
      #pragma unroll
      for (int e = 0; e < 8; ++e) {
        int m = m0 + 16 * i + h * 8 + e;
        if (m < Mvalid) {
          float v = acc[i][j][e];
          if (STORE_MODE == 0)      ((float*)C)[(size_t)m * ldc + n]  = v;
          else if (STORE_MODE == 1) ((bf16_t*)C)[(size_t)m * ldc + n] = (bf16_t)v;
          else                      ((bf16_t*)C)[(size_t)n * ldc + m] = (bf16_t)v;
        }
      }
    }
  }
}

// ---- f32 -> bf16 with row padding (pad rows zeroed) ------------------------
__global__ void convert_pad_bf16(const float* __restrict__ src, bf16_t* __restrict__ dst,
                                 int Mrows, int Mpad, int Ncols, float scale)
{
  int idx = blockIdx.x * blockDim.x + threadIdx.x;
  if (idx >= Mpad * Ncols) return;
  int row = idx / Ncols, col = idx % Ncols;
  float v = (row < Mrows) ? src[(size_t)row * Ncols + col] * scale : 0.f;
  dst[idx] = (bf16_t)v;
}

// ---- f32 (R x C) -> bf16 transposed (C x R), LDS-tiled ---------------------
__global__ __launch_bounds__(256)
void transpose_convert_bf16(const float* __restrict__ src, bf16_t* __restrict__ dst,
                            int R, int C, float scale)
{
  __shared__ float tile[32][33];
  int tx = threadIdx.x, ty = threadIdx.y;     // 32 x 8
  int c0 = blockIdx.x * 32, r0 = blockIdx.y * 32;
  #pragma unroll
  for (int j = 0; j < 32; j += 8)
    tile[ty + j][tx] = src[(size_t)(r0 + ty + j) * C + (c0 + tx)];
  __syncthreads();
  #pragma unroll
  for (int j = 0; j < 32; j += 8)
    dst[(size_t)(c0 + ty + j) * R + (r0 + tx)] = (bf16_t)(tile[tx][ty + j] * scale);
}

// ---- gather relational bias + softmax; emit A (f32) and A (bf16, padded) ---
__global__ __launch_bounds__(256)
void softmax_rel(const float* __restrict__ S, const float* __restrict__ Brel,
                 const int* __restrict__ table,
                 float* __restrict__ Aout, bf16_t* __restrict__ Abf)
{
  __shared__ float red[256];
  int i = blockIdx.x;          // 0..MP-1
  int j = threadIdx.x;         // 0..255
  if (i >= NSEQ) {             // zero pad rows of bf16 A so AV GEMM is exact
    for (int c = j; c < MP; c += 256) Abf[(size_t)i * MP + c] = (bf16_t)0.f;
    return;
  }
  float logit = -3.0e38f;
  if (j < NSEQ) {
    int t = table[(size_t)i * NSEQ + j];
    logit = S[(size_t)i * MP + j] + Brel[(size_t)i * NRELP + t];
  }
  red[j] = logit;
  __syncthreads();
  for (int s = 128; s > 0; s >>= 1) {
    if (j < s) red[j] = fmaxf(red[j], red[j + s]);
    __syncthreads();
  }
  float mx = red[0];
  __syncthreads();
  float e = (j < NSEQ) ? expf(logit - mx) : 0.f;
  red[j] = e;
  __syncthreads();
  for (int s = 128; s > 0; s >>= 1) {
    if (j < s) red[j] += red[j + s];
    __syncthreads();
  }
  float inv = 1.f / red[0];
  if (j < NSEQ) {
    float a = e * inv;
    Aout[(size_t)i * NSEQ + j] = a;
    Abf[(size_t)i * MP + j]    = (bf16_t)a;
  } else if (j < MP) {
    Abf[(size_t)i * MP + j] = (bf16_t)0.f;   // pad cols zeroed
  }
}

// ---------------------------------------------------------------------------
// mode: 0 f32, 1 bf16, 2 bf16-transposed ; bn: 2 (32x32 tile) or 4 (32x64)
static inline void launch_gemm(int mode, int bn,
                               const bf16_t* A, int lda, const bf16_t* Bt, int ldb,
                               void* C, int ldc, int Mrows, int Ncols, int Kdim, int Mvalid,
                               hipStream_t s)
{
  int Mt = Mrows / 32;
  int Nt = Ncols / (16 * bn);
  int blocks = (Mt * Nt + 7) / 8;
  if (bn == 4) {
    if (mode == 0)
      gemm_wmma_bf16<0, 4><<<blocks, 256, 0, s>>>(A, lda, Bt, ldb, C, ldc, Mt, Nt, Kdim, Mvalid);
    else if (mode == 1)
      gemm_wmma_bf16<1, 4><<<blocks, 256, 0, s>>>(A, lda, Bt, ldb, C, ldc, Mt, Nt, Kdim, Mvalid);
    else
      gemm_wmma_bf16<2, 4><<<blocks, 256, 0, s>>>(A, lda, Bt, ldb, C, ldc, Mt, Nt, Kdim, Mvalid);
  } else {
    if (mode == 0)
      gemm_wmma_bf16<0, 2><<<blocks, 256, 0, s>>>(A, lda, Bt, ldb, C, ldc, Mt, Nt, Kdim, Mvalid);
    else if (mode == 1)
      gemm_wmma_bf16<1, 2><<<blocks, 256, 0, s>>>(A, lda, Bt, ldb, C, ldc, Mt, Nt, Kdim, Mvalid);
    else
      gemm_wmma_bf16<2, 2><<<blocks, 256, 0, s>>>(A, lda, Bt, ldb, C, ldc, Mt, Nt, Kdim, Mvalid);
  }
}

extern "C" void kernel_launch(void* const* d_in, const int* in_sizes, int n_in,
                              void* d_out, int out_size, void* d_ws, size_t ws_size,
                              hipStream_t stream)
{
  const float* Xmaj = (const float*)d_in[0];
  const float* Xmin = (const float*)d_in[1];
  const float* Wq   = (const float*)d_in[2];
  const float* Wk   = (const float*)d_in[3];
  const float* Wv   = (const float*)d_in[4];
  const float* Wo   = (const float*)d_in[5];
  const float* wrel = (const float*)d_in[6];
  const int*   tab  = (const int*)d_in[7];
  float* Yout = (float*)d_out;               // 210 x 1024
  float* Aout = Yout + (size_t)NSEQ * DVDIM; // 210 x 210

  char* w = (char*)d_ws;
  size_t off = 0;
  auto take = [&](size_t bytes) -> void* {
    void* p = w + off;
    off = (off + bytes + 255) & ~(size_t)255;
    return p;
  };
  bf16_t* xmaj_b = (bf16_t*)take((size_t)MP * DMDIM * 2);
  bf16_t* xmin_b = (bf16_t*)take((size_t)MP * DMDIM * 2);
  bf16_t* wqT    = (bf16_t*)take((size_t)DKDIM * DMDIM * 2);   // 1024 x 2048, pre-scaled
  bf16_t* wkT    = (bf16_t*)take((size_t)DKDIM * DMDIM * 2);
  bf16_t* wvT    = (bf16_t*)take((size_t)DVDIM * DMDIM * 2);
  bf16_t* woT    = (bf16_t*)take((size_t)DVDIM * DVDIM * 2);
  bf16_t* wrel_b = (bf16_t*)take((size_t)NRELP * DKDIM * 2);
  bf16_t* qs_b   = (bf16_t*)take((size_t)MP * DKDIM * 2);
  bf16_t* k_b    = (bf16_t*)take((size_t)MP * DKDIM * 2);
  bf16_t* vt_b   = (bf16_t*)take((size_t)DVDIM * MP * 2);      // 1024 x 224 (V transposed)
  float*  s_f    = (float*) take((size_t)MP * MP * 4);
  float*  brel_f = (float*) take((size_t)MP * NRELP * 4);
  bf16_t* a_b    = (bf16_t*)take((size_t)MP * MP * 2);
  bf16_t* av_b   = (bf16_t*)take((size_t)MP * DVDIM * 2);
  (void)ws_size; (void)in_sizes; (void)n_in; (void)out_size;

  const float qscale = 0.03125f;   // 1/sqrt(1024), folded into Wq

  // 1) precision conversions (memory bound, ~30MB @ 23.3 TB/s)
  {
    int n1 = MP * DMDIM;
    convert_pad_bf16<<<(n1 + 255) / 256, 256, 0, stream>>>(Xmaj, xmaj_b, NSEQ, MP, DMDIM, 1.f);
    convert_pad_bf16<<<(n1 + 255) / 256, 256, 0, stream>>>(Xmin, xmin_b, NSEQ, MP, DMDIM, 1.f);
    int n2 = NRELP * DKDIM;
    convert_pad_bf16<<<(n2 + 255) / 256, 256, 0, stream>>>(wrel, wrel_b, NREL, NRELP, DKDIM, 1.f);
    dim3 blk(32, 8);
    transpose_convert_bf16<<<dim3(DKDIM / 32, DMDIM / 32), blk, 0, stream>>>(Wq, wqT, DMDIM, DKDIM, qscale);
    transpose_convert_bf16<<<dim3(DKDIM / 32, DMDIM / 32), blk, 0, stream>>>(Wk, wkT, DMDIM, DKDIM, 1.f);
    transpose_convert_bf16<<<dim3(DVDIM / 32, DMDIM / 32), blk, 0, stream>>>(Wv, wvT, DMDIM, DVDIM, 1.f);
    transpose_convert_bf16<<<dim3(DVDIM / 32, DVDIM / 32), blk, 0, stream>>>(Wo, woT, DVDIM, DVDIM, 1.f);
  }

  // 2) projections: Qs, K (row-major bf16), V (stored transposed from epilogue)
  launch_gemm(1, 4, xmaj_b, DMDIM, wqT, DMDIM, qs_b, DKDIM, MP, DKDIM, DMDIM, MP, stream);
  launch_gemm(1, 4, xmin_b, DMDIM, wkT, DMDIM, k_b,  DKDIM, MP, DKDIM, DMDIM, MP, stream);
  launch_gemm(2, 4, xmin_b, DMDIM, wvT, DMDIM, vt_b, MP,    MP, DVDIM, DMDIM, MP, stream);

  // 3) logits: S = Qs K^T ; relational B = Qs wk^T (210x65 GEMM, not 210x210x1024!)
  launch_gemm(0, 2, qs_b, DKDIM, k_b,    DKDIM, s_f,    MP,    MP, MP,    DKDIM, MP, stream);
  launch_gemm(0, 2, qs_b, DKDIM, wrel_b, DKDIM, brel_f, NRELP, MP, NRELP, DKDIM, MP, stream);

  // 4) gather + softmax -> A (f32 output) and A (bf16, zero-padded 224x224)
  softmax_rel<<<MP, 256, 0, stream>>>(s_f, brel_f, tab, Aout, a_b);

  // 5) AV = A V (via V^T), then Y = AV Wo
  launch_gemm(1, 4, a_b,  MP,    vt_b, MP,    av_b, DVDIM, MP, DVDIM, MP,    MP,   stream);
  launch_gemm(0, 4, av_b, DVDIM, woT,  DVDIM, Yout, DVDIM, MP, DVDIM, DVDIM, NSEQ, stream);
}
